// BinaryLinearWscales_61332132987378
// MI455X (gfx1250) — matrix operand order repaired and checked by reading, not compiled, not run on hardware
//
#include <hip/hip_runtime.h>
#include <hip/hip_bf16.h>

// Problem dims (from reference): B=2, S=2048, DIN=4096, DOUT=4096
#define M_TOT 4096   // B*S
#define K_TOT 4096   // DIN
#define N_TOT 4096   // DOUT

typedef _Float16 v4h  __attribute__((ext_vector_type(4)));
typedef _Float16 v8h  __attribute__((ext_vector_type(8)));
typedef _Float16 v16h __attribute__((ext_vector_type(16)));
typedef float    v4f  __attribute__((ext_vector_type(4)));
typedef float    v8f  __attribute__((ext_vector_type(8)));
typedef unsigned int v4u __attribute__((ext_vector_type(4)));
typedef int      v4i  __attribute__((ext_vector_type(4)));
typedef int      v8i  __attribute__((ext_vector_type(8)));

#if __has_builtin(__builtin_amdgcn_tensor_load_to_lds) && __has_builtin(__builtin_amdgcn_s_wait_tensorcnt)
#define USE_TDM 1
#endif
#if defined(USE_TDM) && __has_include(<hip/amd_detail/amd_gfx1250_TDM.h>)
#define TDM_SIX_ARGS 1
#endif

// ---------------------------------------------------------------------------
// Prep 1: wsT[k][n] = (f16) sign(weight[n][k])   (transposed via LDS tile)
// ---------------------------------------------------------------------------
__global__ __launch_bounds__(256) void prep_w_signT(const float* __restrict__ w,
                                                    _Float16* __restrict__ wsT) {
  __shared__ _Float16 tile[32][36];   // padded to dodge bank conflicts
  const int t  = threadIdx.x;
  const int ntn = N_TOT / 32;
  const int tn = blockIdx.x % ntn;    // n-tile
  const int tk = blockIdx.x / ntn;    // k-tile
  const int row = t >> 3;             // 0..31
  const int c4  = (t & 7) * 4;        // 0..28 step 4

  v4f v = *reinterpret_cast<const v4f*>(w + (size_t)(tn * 32 + row) * K_TOT + tk * 32 + c4);
#pragma unroll
  for (int i = 0; i < 4; ++i) {
    float xv = v[i];
    float s  = (xv > 0.f) ? 1.f : ((xv < 0.f) ? -1.f : 0.f);  // jnp.sign semantics
    tile[row][c4 + i] = (_Float16)s;
  }
  __syncthreads();
  v4h o;
#pragma unroll
  for (int i = 0; i < 4; ++i) o[i] = tile[c4 + i][row];       // transpose read
  *reinterpret_cast<v4h*>(wsT + (size_t)(tk * 32 + row) * N_TOT + tn * 32 + c4) = o;
}

// ---------------------------------------------------------------------------
// Prep 2: split x (f32) into x_hi + x_lo (f16 pair, ~22 mantissa bits total)
// ---------------------------------------------------------------------------
__global__ __launch_bounds__(256) void prep_x_split(const float* __restrict__ x,
                                                    _Float16* __restrict__ xhi,
                                                    _Float16* __restrict__ xlo) {
  size_t i = ((size_t)blockIdx.x * 256 + threadIdx.x) * 4;
  v4f v = *reinterpret_cast<const v4f*>(x + i);
  v4h hi, lo;
#pragma unroll
  for (int j = 0; j < 4; ++j) {
    _Float16 h = (_Float16)v[j];
    hi[j] = h;
    lo[j] = (_Float16)(v[j] - (float)h);
  }
  *reinterpret_cast<v4h*>(xhi + i) = hi;
  *reinterpret_cast<v4h*>(xlo + i) = lo;
}

// ---------------------------------------------------------------------------
// Prep 3: f32 row sums of x (for the wbias rank-1 term)
// ---------------------------------------------------------------------------
__global__ __launch_bounds__(256) void rowsum_kernel(const float* __restrict__ x,
                                                     float* __restrict__ rs) {
  __shared__ float red[256];
  const int m = blockIdx.x;
  const int t = threadIdx.x;
  const float* row = x + (size_t)m * K_TOT;
  float s = 0.f;
#pragma unroll
  for (int it = 0; it < 4; ++it) {
    v4f v = *reinterpret_cast<const v4f*>(row + it * 1024 + t * 4);
    s += v[0] + v[1] + v[2] + v[3];
  }
  red[t] = s;
  __syncthreads();
  for (int off = 128; off > 0; off >>= 1) {
    if (t < off) red[t] += red[t + off];
    __syncthreads();
  }
  if (t == 0) rs[m] = red[0];
}

// ---------------------------------------------------------------------------
// TDM helpers: build D# descriptor groups per ISA cdna5 §8.3/8.4 and issue a
// 2D tile load (tile_dim0 contiguous elements per row, tile_dim1 rows, row
// stride in elements). HW adds (`pad_amount`+1) dwords to the LDS address every
// (1<<pad_interval)*2 dwords -> reproduces our padded LDS strides for free.
// ---------------------------------------------------------------------------
#ifdef USE_TDM
__device__ __forceinline__ unsigned lds_off(const void* p) {
  // generic pointer to LDS: low 32 bits are the LDS byte offset
  return (unsigned)(unsigned long long)(uintptr_t)p;
}

__device__ __forceinline__ void tdm_load_2d(const void* gptr, unsigned lds_addr,
                                            unsigned tile_d0, unsigned tile_d1,
                                            unsigned long long stride_elems,
                                            unsigned pad_interval, unsigned pad_amount) {
  unsigned long long ga = (unsigned long long)(uintptr_t)gptr;
  v4u g0;
  g0[0] = 1u;                                   // count=1, user mode, no gather
  g0[1] = lds_addr;                             // LDS byte address
  g0[2] = (unsigned)(ga & 0xFFFFFFFFull);       // global_addr[31:0]
  g0[3] = (unsigned)((ga >> 32) & 0x01FFFFFFull) | (2u << 30);  // [56:32] | type=2

  v8i g1;
  g1[0] = (int)((1u << 16) |                    // data_size = 1 -> 2 bytes
                (1u << 20) |                    // pad_enable
                (pad_interval << 22) |          // pad every (1<<pi)*2 dwords
                (pad_amount << 25));            // pad (pa+1) dwords
  g1[1] = (int)((tile_d0 & 0xFFFFu) << 16);     // tensor_dim0[15:0] (= tile_d0; never OOB)
  g1[2] = (int)((tile_d0 >> 16) | ((tile_d1 & 0xFFFFu) << 16));  // dim0[31:16] | dim1[15:0]
  g1[3] = (int)((tile_d1 >> 16) | (tile_d0 << 16));              // dim1[31:16] | tile_dim0
  g1[4] = (int)(tile_d1 & 0xFFFFu);             // tile_dim1 | tile_dim2=0
  g1[5] = (int)(unsigned)(stride_elems & 0xFFFFFFFFull);         // dim0 stride lo
  g1[6] = (int)(unsigned)((stride_elems >> 32) & 0xFFFFull);     // stride hi | dim1_stride=0
  g1[7] = 0;

  v4i z4 = {0, 0, 0, 0};
#ifdef TDM_SIX_ARGS
  v8i z8 = {0, 0, 0, 0, 0, 0, 0, 0};
  __builtin_amdgcn_tensor_load_to_lds(g0, g1, z4, z4, z8, 0);
#else
  __builtin_amdgcn_tensor_load_to_lds(g0, g1, z4, z4, 0);
#endif
}
#endif  // USE_TDM

// ---------------------------------------------------------------------------
// Main GEMM: out = x @ sign(w)^T via v_wmma_f32_16x16x32_f16 (hi+lo split),
// tiles streamed by the Tensor Data Mover into double-buffered LDS.
// Epilogue: out = wscale[n]*acc + wbias[n]*rowsum[m]
// ---------------------------------------------------------------------------
#define BM 128
#define BN 128
#define BK 32
#define LDA 40    // padded row stride (halves) for A tiles  (64B + 16B pad)
#define LDB 136   // padded row stride (halves) for B tile   (256B + 16B pad)

__global__ __launch_bounds__(256, 1) void binlin_wmma_gemm(
    const _Float16* __restrict__ Ahi, const _Float16* __restrict__ Alo,
    const _Float16* __restrict__ Bt,  // [K_TOT][N_TOT] sign matrix, k-major
    const float* __restrict__ wscale, const float* __restrict__ wbias,
    const float* __restrict__ rowsum, float* __restrict__ out) {
#ifdef USE_TDM
  __shared__ _Float16 sAhi[2][BM * LDA];
  __shared__ _Float16 sAlo[2][BM * LDA];
  __shared__ _Float16 sB[2][BK * LDB];
#else
  __shared__ _Float16 sAhi[1][BM * LDA];
  __shared__ _Float16 sAlo[1][BM * LDA];
  __shared__ _Float16 sB[1][BK * LDB];
#endif

  const int tid  = threadIdx.x;
  const int lane = tid & 31;
  const int wave = tid >> 5;
  const int wm = wave >> 1;  // 0..3  -> wave M offset = wm*32
  const int wn = wave & 1;   // 0..1  -> wave N offset = wn*64

  const int tilen = blockIdx.x & (N_TOT / BN - 1);
  const int tilem = blockIdx.x / (N_TOT / BN);
  const int m0 = tilem * BM;
  const int n0 = tilen * BN;

  v8f acc[2][4];
#pragma unroll
  for (int mi = 0; mi < 2; ++mi)
#pragma unroll
    for (int ni = 0; ni < 4; ++ni) acc[mi][ni] = (v8f){0.f, 0.f, 0.f, 0.f, 0.f, 0.f, 0.f, 0.f};

  // ISA 16-bit A-fragment layout: lane L holds row M=L&15,
  // K runs [base, base+8) and [base+16, base+24), base = (L<16)?0:8
  const int ar  = lane & 15;
  const int akb = (lane >> 4) << 3;

#ifdef USE_TDM
  // -------- TDM double-buffered pipeline --------
  const unsigned oAhi[2] = {lds_off(&sAhi[0][0]), lds_off(&sAhi[1][0])};
  const unsigned oAlo[2] = {lds_off(&sAlo[0][0]), lds_off(&sAlo[1][0])};
  const unsigned oB[2]   = {lds_off(&sB[0][0]), lds_off(&sB[1][0])};

  // prologue: fill buffer 0 (waves 0/1/2 each own one tile stream)
  if (wave == 0) tdm_load_2d(Ahi + (size_t)m0 * K_TOT, oAhi[0], BK, BM, K_TOT, 3u, 3u);
  if (wave == 1) tdm_load_2d(Alo + (size_t)m0 * K_TOT, oAlo[0], BK, BM, K_TOT, 3u, 3u);
  if (wave == 2) tdm_load_2d(Bt + n0, oB[0], BN, BK, N_TOT, 5u, 3u);
  if (wave < 3) __builtin_amdgcn_s_wait_tensorcnt(0);
  __syncthreads();

  for (int kt = 0; kt < K_TOT; kt += BK) {
    const int buf = (kt >> 5) & 1;
    const int nxt = buf ^ 1;
    if (kt + BK < K_TOT) {  // prefetch next tile set via TDM
      const int kn = kt + BK;
      if (wave == 0) tdm_load_2d(Ahi + (size_t)m0 * K_TOT + kn, oAhi[nxt], BK, BM, K_TOT, 3u, 3u);
      if (wave == 1) tdm_load_2d(Alo + (size_t)m0 * K_TOT + kn, oAlo[nxt], BK, BM, K_TOT, 3u, 3u);
      if (wave == 2) tdm_load_2d(Bt + (size_t)kn * N_TOT + n0, oB[nxt], BN, BK, N_TOT, 5u, 3u);
    }
#else
  // -------- fallback: manual staging, single buffer --------
  const int arow = tid >> 1;        // 0..127
  const int acol = (tid & 1) * 16;  // 0 or 16
  const int brow = tid >> 3;        // 0..31
  const int bcol = (tid & 7) * 16;  // 0..112
  const _Float16* gAhi = Ahi + (size_t)(m0 + arow) * K_TOT + acol;
  const _Float16* gAlo = Alo + (size_t)(m0 + arow) * K_TOT + acol;
  const _Float16* gB   = Bt + (size_t)brow * N_TOT + n0 + bcol;

  for (int kt = 0; kt < K_TOT; kt += BK) {
    const int buf = 0;
    v8h a0 = *reinterpret_cast<const v8h*>(gAhi + kt);
    v8h a1 = *reinterpret_cast<const v8h*>(gAhi + kt + 8);
    v8h l0 = *reinterpret_cast<const v8h*>(gAlo + kt);
    v8h l1 = *reinterpret_cast<const v8h*>(gAlo + kt + 8);
    v8h b0 = *reinterpret_cast<const v8h*>(gB + (size_t)kt * N_TOT);
    v8h b1 = *reinterpret_cast<const v8h*>(gB + (size_t)kt * N_TOT + 8);
    __syncthreads();
    *reinterpret_cast<v8h*>(&sAhi[0][arow * LDA + acol])     = a0;
    *reinterpret_cast<v8h*>(&sAhi[0][arow * LDA + acol + 8]) = a1;
    *reinterpret_cast<v8h*>(&sAlo[0][arow * LDA + acol])     = l0;
    *reinterpret_cast<v8h*>(&sAlo[0][arow * LDA + acol + 8]) = l1;
    *reinterpret_cast<v8h*>(&sB[0][brow * LDB + bcol])       = b0;
    *reinterpret_cast<v8h*>(&sB[0][brow * LDB + bcol + 8])   = b1;
    __syncthreads();
#endif

    v16h af[2], lf[2], bf[4];
#pragma unroll
    for (int mi = 0; mi < 2; ++mi) {
      const int rA = (wm * 32 + mi * 16 + ar) * LDA + akb;
      v8h h0 = *reinterpret_cast<const v8h*>(&sAhi[buf][rA]);
      v8h h1 = *reinterpret_cast<const v8h*>(&sAhi[buf][rA + 16]);
      af[mi] = __builtin_shufflevector(h0, h1, 0, 1, 2, 3, 4, 5, 6, 7, 8, 9, 10, 11, 12, 13, 14, 15);
      v8h q0 = *reinterpret_cast<const v8h*>(&sAlo[buf][rA]);
      v8h q1 = *reinterpret_cast<const v8h*>(&sAlo[buf][rA + 16]);
      lf[mi] = __builtin_shufflevector(q0, q1, 0, 1, 2, 3, 4, 5, 6, 7, 8, 9, 10, 11, 12, 13, 14, 15);
    }
#pragma unroll
    for (int ni = 0; ni < 4; ++ni) {
      // ISA B-fragment layout: lane L holds column N=L&15, K in the same order as A
      const int cB = wn * 64 + ni * 16 + ar;
#pragma unroll
      for (int s = 0; s < 16; ++s) {
        const int k = (s < 8) ? (akb + s) : (16 + akb + (s - 8));
        bf[ni][s] = sB[buf][k * LDB + cB];
      }
    }
#pragma unroll
    for (int mi = 0; mi < 2; ++mi)
#pragma unroll
      for (int ni = 0; ni < 4; ++ni) {
        acc[mi][ni] = __builtin_amdgcn_wmma_f32_16x16x32_f16(
            false, af[mi], false, bf[ni], (short)0, acc[mi][ni], false, false);
        acc[mi][ni] = __builtin_amdgcn_wmma_f32_16x16x32_f16(
            false, lf[mi], false, bf[ni], (short)0, acc[mi][ni], false, false);
      }

#ifdef USE_TDM
    if (wave < 3) __builtin_amdgcn_s_wait_tensorcnt(0);  // next buffer landed
    __syncthreads();
  }
#else
  }
#endif

  // epilogue: C/D layout — VGPR r, lane L: N = L&15, M = r + 8*(L>>4)
  const int nBase = n0 + wn * 64;
#pragma unroll
  for (int ni = 0; ni < 4; ++ni) {
    const int n  = nBase + ni * 16 + ar;
    const float wsc = wscale[n];
    const float wbi = wbias[n];
#pragma unroll
    for (int mi = 0; mi < 2; ++mi) {
      const int mb = m0 + wm * 32 + mi * 16 + ((lane >> 4) << 3);
#pragma unroll
      for (int r = 0; r < 8; ++r) {
        const int m = mb + r;
        out[(size_t)m * N_TOT + n] = wsc * acc[mi][ni][r] + wbi * rowsum[m];
      }
    }
  }
}

// ---------------------------------------------------------------------------
extern "C" void kernel_launch(void* const* d_in, const int* in_sizes, int n_in,
                              void* d_out, int out_size, void* d_ws, size_t ws_size,
                              hipStream_t stream) {
  (void)in_sizes; (void)n_in; (void)out_size; (void)ws_size;
  const float* x      = (const float*)d_in[0];   // [2,2048,4096]
  const float* weight = (const float*)d_in[1];   // [4096,4096]
  const float* wscale = (const float*)d_in[2];   // [4096,1]
  const float* wbias  = (const float*)d_in[3];   // [4096,1]
  float* out = (float*)d_out;

  // workspace layout (bytes): wsT 32MB | xhi 32MB | xlo 32MB | rowsum 16KB
  const size_t SZ_HALF_MAT = (size_t)M_TOT * K_TOT * sizeof(_Float16);  // 32MB
  char* ws = (char*)d_ws;
  _Float16* wsT = (_Float16*)(ws);
  _Float16* xhi = (_Float16*)(ws + SZ_HALF_MAT);
  _Float16* xlo = (_Float16*)(ws + 2 * SZ_HALF_MAT);
  float*    rs  = (float*)(ws + 3 * SZ_HALF_MAT);

  prep_w_signT<<<(K_TOT / 32) * (N_TOT / 32), 256, 0, stream>>>(weight, wsT);
  prep_x_split<<<(int)(((size_t)M_TOT * K_TOT) / 1024), 256, 0, stream>>>(x, xhi, xlo);
  rowsum_kernel<<<M_TOT, 256, 0, stream>>>(x, rs);
  binlin_wmma_gemm<<<(M_TOT / BM) * (N_TOT / BN), 256, 0, stream>>>(
      xhi, xlo, wsT, wscale, wbias, rs, out);
}
